// OpVAE_71983651881593
// MI455X (gfx1250) — matrix operand order repaired and checked by reading, not compiled, not run on hardware
//
#include <hip/hip_runtime.h>
#include <hip/hip_bf16.h>

typedef __attribute__((ext_vector_type(16))) __bf16 v16bf;
typedef __attribute__((ext_vector_type(4)))  __bf16 v4bf;
typedef __attribute__((ext_vector_type(8)))  float  v8f;
typedef __attribute__((ext_vector_type(4)))  float  v4f;

#define TILE_M 128
#define TILE_N 64
#define TILE_K 32

// ---------------------------------------------------------------------------
// Generic WMMA GEMM: C[M,N] = A1 . B1^T + A2 . B2^T + bias1 + bias2
//   A matrices: fp32, row-major, row stride ld; optional row-index gather
//   B matrices: fp32, [N x K] row-major (row stride == K)
//   Block tile 128x64, 8 waves, each wave a 32x32 sub-tile (2x2 WMMA frags).
//   Staging: float4 global loads -> v_cvt_pk_bf16_f32 -> b64 LDS stores;
//   row pointers (incl. embedding gather) resolved once per pass.
//   Math: v_wmma_f32_16x16x32_bf16 with f32 accumulation.
// ---------------------------------------------------------------------------
__global__ __launch_bounds__(256)
void wmma_gemm_kernel(const float* __restrict__ A1, long lda1,
                      const int*  __restrict__ aidx, int aidx_stride,
                      const float* __restrict__ A2, long lda2,
                      const float* __restrict__ B1, const float* __restrict__ B2,
                      int K1, int K2,
                      const float* __restrict__ bias1, const float* __restrict__ bias2,
                      float* __restrict__ C, long ldc, int M, int N)
{
    __shared__ __bf16 sA[TILE_M * TILE_K];   // 8 KB
    __shared__ __bf16 sB[TILE_N * TILE_K];   // 4 KB

    const int tid  = threadIdx.x;
    const int lane = tid & 31;
    const int wave = tid >> 5;
    const int wm   = wave & 3;   // 4 wave-rows  -> 32-row strip each
    const int wn   = wave >> 2;  // 2 wave-cols  -> 32-col strip each
    const long rowBase = (long)blockIdx.y * TILE_M;
    const long colBase = (long)blockIdx.x * TILE_N;

    // staging assignment: 8 threads per row, 4 consecutive floats each
    const int rL = tid >> 3;          // 0..31 (+32h for further strips)
    const int k4 = (tid & 7) << 2;    // 0,4,...,28

    v8f acc00 = {}, acc01 = {}, acc10 = {}, acc11 = {};

    for (int pass = 0; pass < 2; ++pass) {
        const float* Ap = pass ? A2 : A1;
        const float* Bp = pass ? B2 : B1;
        const int    Kp = pass ? K2 : K1;
        const long   ld = pass ? lda2 : lda1;
        const int*  idx = pass ? nullptr : aidx;
        if (Ap == nullptr || Bp == nullptr || Kp <= 0) continue;

        // ---- resolve row pointers once per pass (K-invariant) -------------
        const float* aptr[4];
        const float* bptr[2];
        #pragma unroll
        for (int h = 0; h < 4; ++h) {
            long gr = rowBase + rL + h * 32;
            aptr[h] = idx ? (Ap + (long)idx[gr * (long)aidx_stride] * ld)
                          : (Ap + gr * ld);
        }
        #pragma unroll
        for (int h = 0; h < 2; ++h) {
            bptr[h] = Bp + (colBase + rL + h * 32) * (long)Kp;
        }

        for (int k0 = 0; k0 < Kp; k0 += TILE_K) {
            // ---- stage A (128x32) and B (64x32), fp32 -> bf16, vectorized -
            #pragma unroll
            for (int h = 0; h < 4; ++h) {
                v4f av = *(const v4f*)(aptr[h] + k0 + k4);
                v4bf apk;
                #pragma unroll
                for (int j = 0; j < 4; ++j) apk[j] = (__bf16)av[j];
                *(v4bf*)&sA[(rL + h * 32) * TILE_K + k4] = apk;
            }
            #pragma unroll
            for (int h = 0; h < 2; ++h) {
                v4f bv = *(const v4f*)(bptr[h] + k0 + k4);
                v4bf bpk;
                #pragma unroll
                for (int j = 0; j < 4; ++j) bpk[j] = (__bf16)bv[j];
                *(v4bf*)&sB[(rL + h * 32) * TILE_K + k4] = bpk;
            }
            // prefetch next K-chunk into near caches (WGP-scope; L2-hot data)
            if (k0 + TILE_K < Kp) {
                __builtin_prefetch(aptr[0] + k0 + TILE_K + k4, 0, 3);
                __builtin_prefetch(bptr[0] + k0 + TILE_K + k4, 0, 3);
            }
            __syncthreads();

            // ---- build fragments (documented CDNA5 16-bit layouts) --------
            const int half = lane >> 4;
            v16bf af0, af1;
            {
                const int m = lane & 15;
                const __bf16* a0 = &sA[(wm * 32 + m) * TILE_K];
                const __bf16* a1 = &sA[(wm * 32 + 16 + m) * TILE_K];
                #pragma unroll
                for (int j = 0; j < 16; ++j) {
                    int k = ((j < 8) ? j : (j + 8)) + half * 8;
                    af0[j] = a0[k];
                    af1[j] = a1[k];
                }
            }
            v16bf bf0, bf1;
            {
                const int n = lane & 15;
                const __bf16* b0 = &sB[(wn * 32 + n) * TILE_K];
                const __bf16* b1 = &sB[(wn * 32 + 16 + n) * TILE_K];
                #pragma unroll
                for (int j = 0; j < 16; ++j) {
                    int k = ((j < 8) ? j : (j + 8)) + half * 8;
                    bf0[j] = b0[k];
                    bf1[j] = b1[k];
                }
            }

            acc00 = __builtin_amdgcn_wmma_f32_16x16x32_bf16(
                        false, af0, false, bf0, (short)0, acc00, false, false);
            acc01 = __builtin_amdgcn_wmma_f32_16x16x32_bf16(
                        false, af0, false, bf1, (short)0, acc01, false, false);
            acc10 = __builtin_amdgcn_wmma_f32_16x16x32_bf16(
                        false, af1, false, bf0, (short)0, acc10, false, false);
            acc11 = __builtin_amdgcn_wmma_f32_16x16x32_bf16(
                        false, af1, false, bf1, (short)0, acc11, false, false);
            __syncthreads();
        }
    }

    // ---- writeback with fused (dual) bias ---------------------------------
    const int n    = lane & 15;
    const int half = lane >> 4;
    long c0 = colBase + wn * 32 + n;
    long c1 = c0 + 16;
    float b0 = (bias1 ? bias1[c0] : 0.f) + (bias2 ? bias2[c0] : 0.f);
    float b1 = (bias1 ? bias1[c1] : 0.f) + (bias2 ? bias2[c1] : 0.f);
    #pragma unroll
    for (int r = 0; r < 8; ++r) {
        int  m    = r + half * 8;
        long row0 = rowBase + wm * 32 + m;
        long row1 = row0 + 16;
        C[row0 * ldc + c0] = acc00[r] + b0;
        C[row0 * ldc + c1] = acc01[r] + b1;
        C[row1 * ldc + c0] = acc10[r] + b0;
        C[row1 * ldc + c1] = acc11[r] + b1;
    }
}

// ---------------------------------------------------------------------------
// LSTM cell pointwise update: c' = sig(f)*c + sig(i)*tanh(g); h' = sig(o)*tanh(c')
// g layout: [B, 4H] with gate order i,f,g,o. Optionally stores h into outs[(b*T+t)*H+j].
// ---------------------------------------------------------------------------
__global__ void lstm_cell_kernel(const float* __restrict__ g,
                                 float* __restrict__ c, float* __restrict__ h,
                                 float* __restrict__ outs, int t, int T,
                                 int Bn, int Hn)
{
    long i = (long)blockIdx.x * blockDim.x + threadIdx.x;
    long total = (long)Bn * Hn;
    if (i >= total) return;
    long b = i / Hn;
    long j = i - b * Hn;
    const float* gr = g + b * (4L * Hn);
    float gi = gr[j];
    float gf = gr[Hn + j];
    float gg = gr[2L * Hn + j];
    float go = gr[3L * Hn + j];
    float si = 1.f / (1.f + expf(-gi));
    float sf = 1.f / (1.f + expf(-gf));
    float so = 1.f / (1.f + expf(-go));
    float cn = sf * c[i] + si * tanhf(gg);
    float hn = so * tanhf(cn);
    c[i] = cn;
    h[i] = hn;
    if (outs) outs[(b * T + t) * (long)Hn + j] = hn;
}

// ---------------------------------------------------------------------------
// LayerNorm over concat(hf[b], hb[b]) (2H = 1024), two parameter sets.
// One block per batch row; 256 threads, 4 elements each.
// ---------------------------------------------------------------------------
__global__ __launch_bounds__(256)
void ln_heads_kernel(const float* __restrict__ hf, const float* __restrict__ hb,
                     const float* __restrict__ mg, const float* __restrict__ mb,
                     const float* __restrict__ lg, const float* __restrict__ lb,
                     float* __restrict__ ln_mu, float* __restrict__ ln_lv, int Hn)
{
    __shared__ float r1[256];
    __shared__ float r2[256];
    const int b = blockIdx.x;
    const int tid = threadIdx.x;
    float vals[4];
    float s = 0.f, s2 = 0.f;
    #pragma unroll
    for (int i = 0; i < 4; ++i) {
        int k = tid + i * 256;
        float v = (k < Hn) ? hf[(long)b * Hn + k] : hb[(long)b * Hn + (k - Hn)];
        vals[i] = v;
        s += v; s2 += v * v;
    }
    r1[tid] = s; r2[tid] = s2;
    __syncthreads();
    for (int off = 128; off; off >>= 1) {
        if (tid < off) { r1[tid] += r1[tid + off]; r2[tid] += r2[tid + off]; }
        __syncthreads();
    }
    float mean = r1[0] / (2.f * Hn);
    float var  = r2[0] / (2.f * Hn) - mean * mean;
    float inv  = rsqrtf(var + 1e-5f);
    #pragma unroll
    for (int i = 0; i < 4; ++i) {
        int k = tid + i * 256;
        float xn = (vals[i] - mean) * inv;
        ln_mu[(long)b * (2L * Hn) + k] = xn * mg[k] + mb[k];
        ln_lv[(long)b * (2L * Hn) + k] = xn * lg[k] + lb[k];
    }
}

// ---------------------------------------------------------------------------
// Reparameterize: z = mu + exp(0.5*logvar)*eps ; decoder h0 = c0 = z
// ---------------------------------------------------------------------------
__global__ void z_init_kernel(const float* __restrict__ mu, const float* __restrict__ lv,
                              const float* __restrict__ eps,
                              float* __restrict__ hd, float* __restrict__ cd, long n)
{
    long i = (long)blockIdx.x * blockDim.x + threadIdx.x;
    if (i >= n) return;
    float z = mu[i] + expf(0.5f * lv[i]) * eps[i];
    hd[i] = z;
    cd[i] = z;
}

__global__ void fill_zero_kernel(float* __restrict__ p, long n)
{
    long i = (long)blockIdx.x * blockDim.x + threadIdx.x;
    if (i < n) p[i] = 0.f;
}

__global__ void copy_feat_kernel(const int* __restrict__ f, float* __restrict__ o, long n)
{
    long i = (long)blockIdx.x * blockDim.x + threadIdx.x;
    if (i < n) o[i] = (float)f[i];
}

// ---------------------------------------------------------------------------
extern "C" void kernel_launch(void* const* d_in, const int* in_sizes, int n_in,
                              void* d_out, int out_size, void* d_ws, size_t ws_size,
                              hipStream_t stream)
{
    const int B = 2048, L = 32, E = 256, H = 512, V = 1024, T = 7;

    const int*   features  = (const int*)  d_in[0];
    const float* eps       = (const float*)d_in[1];
    const float* emb       = (const float*)d_in[2];
    const float* eWih_f    = (const float*)d_in[3];
    const float* eWhh_f    = (const float*)d_in[4];
    const float* ebih_f    = (const float*)d_in[5];
    const float* ebhh_f    = (const float*)d_in[6];
    const float* eWih_b    = (const float*)d_in[7];
    const float* eWhh_b    = (const float*)d_in[8];
    const float* ebih_b    = (const float*)d_in[9];
    const float* ebhh_b    = (const float*)d_in[10];
    /* d_in[11] = dec_Wih : unused by reference (decoder input is bias-only) */
    const float* dWhh      = (const float*)d_in[12];
    const float* dbih      = (const float*)d_in[13];
    const float* dbhh      = (const float*)d_in[14];
    const float* mu_ln_g   = (const float*)d_in[15];
    const float* mu_ln_b   = (const float*)d_in[16];
    const float* mu_W      = (const float*)d_in[17];
    const float* mu_b      = (const float*)d_in[18];
    const float* lv_ln_g   = (const float*)d_in[19];
    const float* lv_ln_b   = (const float*)d_in[20];
    const float* lv_W      = (const float*)d_in[21];
    const float* lv_b      = (const float*)d_in[22];
    const float* ff_W      = (const float*)d_in[23];
    const float* ff_b      = (const float*)d_in[24];

    // ---- workspace layout (floats) ----------------------------------------
    float* ws = (float*)d_ws;
    float* g     = ws;                 ws += (size_t)B * 4 * H;   // gate buffer
    float* hf    = ws;                 ws += (size_t)B * H;       // fwd h
    float* cf    = ws;                 ws += (size_t)B * H;       // fwd c
    float* hb    = ws;                 ws += (size_t)B * H;       // bwd h
    float* cb    = ws;                 ws += (size_t)B * H;       // bwd c
    float* hd    = ws;                 ws += (size_t)B * H;       // dec h
    float* cd    = ws;                 ws += (size_t)B * H;       // dec c
    float* ln_mu = ws;                 ws += (size_t)B * 2 * H;
    float* ln_lv = ws;                 ws += (size_t)B * 2 * H;
    float* outs  = ws;                 ws += (size_t)B * T * H;   // (B,T,H)

    // ---- output layout -----------------------------------------------------
    float* out_feat = (float*)d_out;
    float* out_fhat = out_feat + (size_t)B * L;
    float* out_mu   = out_fhat + (size_t)B * T * V;
    float* out_lv   = out_mu   + (size_t)B * H;

    auto gemm = [&](const float* A1, long lda1, const int* aidx, int aidxs,
                    const float* A2, long lda2,
                    const float* B1, const float* B2, int K1, int K2,
                    const float* b1, const float* b2,
                    float* C, long ldc, int M, int N) {
        dim3 grid(N / TILE_N, M / TILE_M);
        wmma_gemm_kernel<<<grid, 256, 0, stream>>>(A1, lda1, aidx, aidxs,
                                                   A2, lda2, B1, B2, K1, K2,
                                                   b1, b2, C, ldc, M, N);
    };

    const long BH = (long)B * H;

    // zero fwd/bwd LSTM states (hf,cf,hb,cb are contiguous)
    fill_zero_kernel<<<(unsigned)((4 * BH + 255) / 256), 256, 0, stream>>>(hf, 4 * BH);
    // features passthrough (int -> float)
    copy_feat_kernel<<<(unsigned)(((long)B * L + 255) / 256), 256, 0, stream>>>(
        features, out_feat, (long)B * L);

    // ---- forward encoder LSTM ---------------------------------------------
    for (int t = 0; t < L; ++t) {
        // gates = emb[features[:,t]] . Wih^T + h . Whh^T + bih + bhh
        gemm(emb, E, features + t, L, hf, H,
             eWih_f, eWhh_f, E, H, ebih_f, ebhh_f, g, 4 * H, B, 4 * H);
        lstm_cell_kernel<<<(unsigned)((BH + 255) / 256), 256, 0, stream>>>(
            g, cf, hf, nullptr, 0, 0, B, H);
    }

    // ---- backward encoder LSTM --------------------------------------------
    for (int t = L - 1; t >= 0; --t) {
        gemm(emb, E, features + t, L, hb, H,
             eWih_b, eWhh_b, E, H, ebih_b, ebhh_b, g, 4 * H, B, 4 * H);
        lstm_cell_kernel<<<(unsigned)((BH + 255) / 256), 256, 0, stream>>>(
            g, cb, hb, nullptr, 0, 0, B, H);
    }

    // ---- LayerNorm heads ---------------------------------------------------
    ln_heads_kernel<<<B, 256, 0, stream>>>(hf, hb, mu_ln_g, mu_ln_b,
                                           lv_ln_g, lv_ln_b, ln_mu, ln_lv, H);
    gemm(ln_mu, 2 * H, nullptr, 0, nullptr, 0, mu_W, nullptr, 2 * H, 0,
         mu_b, nullptr, out_mu, H, B, H);
    gemm(ln_lv, 2 * H, nullptr, 0, nullptr, 0, lv_W, nullptr, 2 * H, 0,
         lv_b, nullptr, out_lv, H, B, H);

    // ---- reparameterize ----------------------------------------------------
    z_init_kernel<<<(unsigned)((BH + 255) / 256), 256, 0, stream>>>(
        out_mu, out_lv, eps, hd, cd, BH);

    // ---- decoder LSTM (input term is bias-only) ---------------------------
    for (int t = 0; t < T; ++t) {
        gemm(hd, H, nullptr, 0, nullptr, 0, dWhh, nullptr, H, 0,
             dbih, dbhh, g, 4 * H, B, 4 * H);
        lstm_cell_kernel<<<(unsigned)((BH + 255) / 256), 256, 0, stream>>>(
            g, cd, hd, outs, t, T, B, H);
    }

    // ---- output projection: (B*T, H) x (V, H)^T ---------------------------
    gemm(outs, H, nullptr, 0, nullptr, 0, ff_W, nullptr, H, 0,
         ff_b, nullptr, out_fhat, V, B * T, V);
}
